// QuantumEntanglementLayer_45569603010751
// MI455X (gfx1250) — matrix operand order repaired and checked by reading, not compiled, not run on hardware
//
#include <hip/hip_runtime.h>
#include <math.h>

// ---------------------------------------------------------------------------
// Types for CDNA5 WMMA (wave32): v_wmma_f32_16x16x32_bf16
// ---------------------------------------------------------------------------
typedef __attribute__((ext_vector_type(16))) __bf16 bf16x16;
typedef __attribute__((ext_vector_type(8)))  float  floatx8;

union FragU {
    uint4   u[2];   // 32 bytes
    bf16x16 v;      // 16 bf16
};

__device__ __forceinline__ unsigned short f2bf_bits(float f) {
    // round-to-nearest-even f32 -> bf16
    unsigned u = __float_as_uint(f);
    unsigned r = u + 0x7FFFu + ((u >> 16) & 1u);
    return (unsigned short)(r >> 16);
}

// ---------------------------------------------------------------------------
// Async global -> LDS (ASYNCcnt path), via inline asm.  The LDS byte offset
// is obtained by C-style cast to an addrspace(3) pointer (the sanctioned way
// to addrspace-cast in clang) + ptrtoint.  The ptrtoint CAPTURES the shared
// array, so the "memory" clobber makes LLVM assume the asm writes it -- the
// subsequent ds_load_b128 fragment reads cannot be folded to undef.
// ---------------------------------------------------------------------------
typedef __attribute__((address_space(3))) unsigned char lds_u8;

__device__ __forceinline__ void async_load_b128(const void* g, void* l) {
    const unsigned off = (unsigned)(unsigned long long)(lds_u8*)l;
    asm volatile("global_load_async_to_lds_b128 %0, %1, off"
                 :: "v"(off), "v"((unsigned long long)g) : "memory");
}

#define WAIT_ASYNC_8() asm volatile("s_wait_asynccnt 0x8" ::: "memory")
#define WAIT_ASYNC_0() asm volatile("s_wait_asynccnt 0x0" ::: "memory")

// ---------------------------------------------------------------------------
// Kernel 1: row softmax of E -> bf16 W, row-major.  One block per row.
// ---------------------------------------------------------------------------
__global__ void __launch_bounds__(256) softmax_bf16_kernel(
    const float* __restrict__ E, unsigned short* __restrict__ Wbf, int N) {
    __shared__ float red[256];
    const int row = blockIdx.x;
    const int t   = threadIdx.x;
    const float* e_row = E + (size_t)row * N;
    const int per = N >> 8;          // N / 256 elements per thread (N=2048 -> 8)
    float e[16];

    float mx = -3.402823e38f;
    for (int i = 0; i < per; ++i) {
        e[i] = e_row[t + (i << 8)];
        mx = fmaxf(mx, e[i]);
    }
    red[t] = mx; __syncthreads();
    for (int s = 128; s > 0; s >>= 1) {
        if (t < s) red[t] = fmaxf(red[t], red[t + s]);
        __syncthreads();
    }
    mx = red[0]; __syncthreads();

    float sum = 0.f;
    for (int i = 0; i < per; ++i) {
        e[i] = __expf(e[i] - mx);
        sum += e[i];
    }
    red[t] = sum; __syncthreads();
    for (int s = 128; s > 0; s >>= 1) {
        if (t < s) red[t] += red[t + s];
        __syncthreads();
    }
    const float inv = 1.f / red[0];

    unsigned short* w_row = Wbf + (size_t)row * N;
    for (int i = 0; i < per; ++i)
        w_row[t + (i << 8)] = f2bf_bits(e[i] * inv);
}

// ---------------------------------------------------------------------------
// Kernel 2: bell (block-diag 2x2) -> bf16 [B,N].  One thread per pair.
// Also zeroes the bell_corr accumulator slot out[B*N].
// ---------------------------------------------------------------------------
__global__ void __launch_bounds__(256) bell_bf16_kernel(
    const float* __restrict__ spikes, const float* __restrict__ Wb,
    unsigned short* __restrict__ bellbf, int total_pairs, int P,
    float* __restrict__ corr_slot) {
    const int idx = blockIdx.x * blockDim.x + threadIdx.x;
    if (idx == 0) *corr_slot = 0.f;
    if (idx >= total_pairs) return;
    const int p = idx & (P - 1);                       // P is a power of two
    const float2 s = reinterpret_cast<const float2*>(spikes)[idx];
    const float4 w = reinterpret_cast<const float4*>(Wb)[p];   // [w00 w01 w10 w11]
    const float y0 = w.x * s.x + w.y * s.y;
    const float y1 = w.z * s.x + w.w * s.y;
    const unsigned pack = (unsigned)f2bf_bits(y0) | ((unsigned)f2bf_bits(y1) << 16);
    reinterpret_cast<unsigned*>(bellbf)[idx] = pack;
}

// ---------------------------------------------------------------------------
// Kernel 3: nonlocal = bell @ W^T via v_wmma_f32_16x16x32_bf16.
//
// Per iteration the WG stages a 128x64 bf16 A-tile (bell) and a 128x64 bf16
// B-tile (W rows) into LDS with global_load_async_to_lds_b128 (ASYNCcnt,
// double-buffered), then runs 2 k-sub-steps x 8 WMMAs from LDS.
// LDS pitch 144B (36 words, gcd(36,64)=4) -> fragment b128 reads cover all
// 64 banks (optimal 2-pass for 512B/wave).
//
// Fragment layouts (CDNA5 ISA §7.12.2, wave32):
//  A (16x32 bf16): lane L<16 -> row L, K {0..15} (contiguous k, same order as B);
//                  lane L>=16 -> row L-16, K {16..31}
//    => two 16B chunks at (row, kbase), (row, kbase+8), kbase=(L>=16)*16.
//  B (32x16 bf16) with B[k,n] = W[n,k]: lane L holds col n=L&15,
//    16 contiguous K starting at (L>=16)*16 => 32B run of W row n.
//  C/D (16x16 f32): VGPR v, lane L -> M = v + 8*(L>=16), Ncol = L&15.
// ---------------------------------------------------------------------------
#define KSTEP   64
#define PITCH   144                   // bytes per LDS tile row (64 bf16 + 16B pad)
#define TILEBYT (128 * PITCH)         // 18432 B per tile buffer
#define LDS_A0  0
#define LDS_B0  (2 * TILEBYT)         // A double-buffer first, then B
#define LDS_TOT (4 * TILEBYT)         // 72 KB

__global__ void __launch_bounds__(256) entangle_gemm_kernel(
    const unsigned short* __restrict__ bellbf,   // [B,N] bf16
    const unsigned short* __restrict__ Wbf,      // [N,N] bf16 softmax rows
    const float* __restrict__ spikes,            // [B,N] f32
    const float* __restrict__ Wb,                // [N/2,2,2] f32
    float* __restrict__ out, int N) {
    __shared__ __align__(16) unsigned char lds[LDS_TOT];

    const int t    = threadIdx.x;
    const int lane = t & 31;
    const int wave = t >> 5;
    const int wm = wave >> 1;                    // 0..3 -> 32 rows each
    const int wn = wave & 1;                     // 0..1 -> 64 cols each
    const int block_m = blockIdx.y * 128;
    const int block_n = blockIdx.x * 128;
    const int l15 = lane & 15;
    const int lhi = lane >> 4;                   // 0 or 1

    // ---- async-fill assignment: thread t -> rows r0+32q (q=0..3), seg s0 ----
    const int r0 = t >> 3;                       // 0..31
    const int s0 = t & 7;                        // 16B segment within 128B row
    unsigned ldsA[4], ldsB[4];
    const unsigned short* gA[4];
    const unsigned short* gB[4];
#pragma unroll
    for (int q = 0; q < 4; ++q) {
        const int r = r0 + 32 * q;
        ldsA[q] = LDS_A0 + (unsigned)(r * PITCH + s0 * 16);
        ldsB[q] = LDS_B0 + (unsigned)(r * PITCH + s0 * 16);
        gA[q] = bellbf + (size_t)(block_m + r) * N + s0 * 8;
        gB[q] = Wbf    + (size_t)(block_n + r) * N + s0 * 8;
    }

    // ---- fragment LDS offsets (constant per lane) ----
    unsigned aoff[2], boff[4];
#pragma unroll
    for (int tm = 0; tm < 2; ++tm)
        aoff[tm] = LDS_A0 + (unsigned)((wm * 32 + tm * 16 + l15) * PITCH + lhi * 32);
#pragma unroll
    for (int tn = 0; tn < 4; ++tn)
        boff[tn] = LDS_B0 + (unsigned)((wn * 64 + tn * 16 + l15) * PITCH + lhi * 32);

    floatx8 acc[2][4];
#pragma unroll
    for (int tm = 0; tm < 2; ++tm)
#pragma unroll
        for (int tn = 0; tn < 4; ++tn)
            acc[tm][tn] = (floatx8)0.f;

    auto issue_fill = [&](unsigned bufoff, int k0) {
#pragma unroll
        for (int q = 0; q < 4; ++q)
            async_load_b128(gA[q] + k0, lds + (ldsA[q] + bufoff));
#pragma unroll
        for (int q = 0; q < 4; ++q)
            async_load_b128(gB[q] + k0, lds + (ldsB[q] + bufoff));
    };

    const int S = N / KSTEP;                     // 32 iterations
    issue_fill(0u, 0);

    for (int i = 0; i < S; ++i) {
        const unsigned curoff = (unsigned)(i & 1) * (unsigned)TILEBYT;
        if (i + 1 < S) {
            issue_fill((unsigned)((i + 1) & 1) * (unsigned)TILEBYT, (i + 1) * KSTEP);
            WAIT_ASYNC_8();                      // current buffer's 8 are done
        } else {
            WAIT_ASYNC_0();
        }
        __syncthreads();

#pragma unroll
        for (int ksub = 0; ksub < 2; ++ksub) {
            const unsigned ks = curoff + (unsigned)(ksub * 64);  // +32 K = +64B
            FragU a[2], b[4];
#pragma unroll
            for (int tm = 0; tm < 2; ++tm) {
                const unsigned char* pa = lds + (ks + aoff[tm]);
                a[tm].u[0] = *(const uint4*)(pa);
                a[tm].u[1] = *(const uint4*)(pa + 16);   // K += 8
            }
#pragma unroll
            for (int tn = 0; tn < 4; ++tn) {
                const unsigned char* pb = lds + (ks + boff[tn]);
                b[tn].u[0] = *(const uint4*)(pb);
                b[tn].u[1] = *(const uint4*)(pb + 16);
            }
#pragma unroll
            for (int tm = 0; tm < 2; ++tm)
#pragma unroll
                for (int tn = 0; tn < 4; ++tn)
                    acc[tm][tn] = __builtin_amdgcn_wmma_f32_16x16x32_bf16(
                        false, a[tm].v, false, b[tn].v,
                        (short)0, acc[tm][tn], false, false);
        }

        __syncthreads();   // all reads done before this buffer is refilled
    }

    // Fused epilogue: recompute bell in exact f32, mix, clip, store.
    const float2* sp2 = reinterpret_cast<const float2*>(spikes);
    const int halfN = N >> 1;
    const int mrow_hi = lhi * 8;
#pragma unroll
    for (int tn = 0; tn < 4; ++tn) {
        const int j  = block_n + wn * 64 + tn * 16 + l15;  // fixed per lane
        const int p  = j >> 1;
        const int jj = j & 1;
        const float wb0 = Wb[p * 4 + jj * 2 + 0];
        const float wb1 = Wb[p * 4 + jj * 2 + 1];
#pragma unroll
        for (int tm = 0; tm < 2; ++tm) {
            const int rbase = block_m + wm * 32 + tm * 16 + mrow_hi;
#pragma unroll
            for (int v = 0; v < 8; ++v) {
                const int row = rbase + v;
                const float2 s = sp2[(size_t)row * halfN + p];
                const float bell = wb0 * s.x + wb1 * s.y;
                float val = 0.5f * bell + 0.5f * acc[tm][tn][v];
                val = fminf(fmaxf(val, 0.f), 1.f);
                out[(size_t)row * N + j] = val;
            }
        }
    }
}

// ---------------------------------------------------------------------------
// Kernel 4: per-row cosine similarity, mean of |corr| via atomicAdd.
// One block per batch row.
// ---------------------------------------------------------------------------
__global__ void __launch_bounds__(256) corr_kernel(
    const float* __restrict__ spikes, float* __restrict__ out, int B, int N) {
    __shared__ float red[5][256];
    const int b = blockIdx.x;
    const int t = threadIdx.x;
    const float4* s4 = reinterpret_cast<const float4*>(spikes + (size_t)b * N);
    const float4* e4 = reinterpret_cast<const float4*>(out + (size_t)b * N);
    const int quads = N >> 2;

    float ss = 0.f, se = 0.f, dt = 0.f, s2 = 0.f, e2 = 0.f;
    for (int i = t; i < quads; i += 256) {
        const float4 s = s4[i];
        const float4 e = e4[i];
        ss += s.x + s.y + s.z + s.w;
        se += e.x + e.y + e.z + e.w;
        dt += s.x * e.x + s.y * e.y + s.z * e.z + s.w * e.w;
        s2 += s.x * s.x + s.y * s.y + s.z * s.z + s.w * s.w;
        e2 += e.x * e.x + e.y * e.y + e.z * e.z + e.w * e.w;
    }
    red[0][t] = ss; red[1][t] = se; red[2][t] = dt; red[3][t] = s2; red[4][t] = e2;
    __syncthreads();
    for (int s = 128; s > 0; s >>= 1) {
        if (t < s)
            for (int q = 0; q < 5; ++q) red[q][t] += red[q][t + s];
        __syncthreads();
    }
    if (t == 0) {
        const float fn = (float)N;
        ss = red[0][0]; se = red[1][0]; dt = red[2][0]; s2 = red[3][0]; e2 = red[4][0];
        const float dc = dt - ss * se / fn;
        float na = sqrtf(fmaxf(s2 - ss * ss / fn, 0.f));
        float nb = sqrtf(fmaxf(e2 - se * se / fn, 0.f));
        na = fmaxf(na, 1e-8f);
        nb = fmaxf(nb, 1e-8f);
        atomicAdd(&out[(size_t)B * N], fabsf(dc / (na * nb)) * (1.f / (float)B));
    }
}

// ---------------------------------------------------------------------------
// Launch
// ---------------------------------------------------------------------------
extern "C" void kernel_launch(void* const* d_in, const int* in_sizes, int n_in,
                              void* d_out, int out_size, void* d_ws, size_t ws_size,
                              hipStream_t stream) {
    const float* spikes = (const float*)d_in[0];   // [B,N]
    const float* Wbell  = (const float*)d_in[1];   // [N/2,2,2] = 2N floats
    const float* E      = (const float*)d_in[2];   // [N,N]
    float* out = (float*)d_out;                    // B*N entangled + 1 scalar

    const int N = in_sizes[1] / 2;                 // W_bell has 2N elements
    const int B = in_sizes[0] / N;

    unsigned short* Wbf    = (unsigned short*)d_ws;                 // N*N bf16 (8 MB)
    unsigned short* bellbf = Wbf + (size_t)N * N;                   // B*N bf16 (64 MB)

    // 1. W = softmax(E) rows, stored bf16 row-major
    softmax_bf16_kernel<<<N, 256, 0, stream>>>(E, Wbf, N);

    // 2. bell in bf16 (+ zero the scalar accumulator)
    const int P = N / 2;
    const int total_pairs = B * P;
    bell_bf16_kernel<<<(total_pairs + 255) / 256, 256, 0, stream>>>(
        spikes, Wbell, bellbf, total_pairs, P, out + (size_t)B * N);

    // 3. WMMA GEMM (async global->LDS double-buffered) + fused epilogue
    dim3 grid(N / 128, B / 128);
    entangle_gemm_kernel<<<grid, 256, 0, stream>>>(bellbf, Wbf, spikes, Wbell, out, N);

    // 4. bell_corr reduction
    corr_kernel<<<B, 256, 0, stream>>>(spikes, out, B, N);
}
